// LeNet_70695161692193
// MI455X (gfx1250) — 8-bit matrix operand order repaired and checked by reading, compiled, not run on hardware
//
#include <hip/hip_runtime.h>
#include <stdint.h>
#include <math.h>


typedef int v8i __attribute__((ext_vector_type(8)));

#define BNN_EPS 1e-5f

__global__ void bnn_prep_affine(const float* __restrict__ bia, const float* __restrict__ g,
                                const float* __restrict__ be, const float* __restrict__ m,
                                const float* __restrict__ v, float* __restrict__ scale,
                                float* __restrict__ shift, int n) {
    int i = blockIdx.x * blockDim.x + threadIdx.x;
    if (i < n) {
        float s = g[i] * rsqrtf(v[i] + BNN_EPS);
        scale[i] = s;
        shift[i] = (bia[i] - m[i]) * s + be[i];
    }
}

__global__ void bnn_pack_sign(const float* __restrict__ w, int8_t* __restrict__ out, int n) {
    int i = blockIdx.x * blockDim.x + threadIdx.x;
    if (i < n) out[i] = (w[i] >= 0.0f) ? (int8_t)1 : (int8_t)-1;
}

__global__ void bnn_pack_w2(const float* __restrict__ w2, int8_t* __restrict__ wb2) {
    int idx = blockIdx.x * blockDim.x + threadIdx.x;
    if (idx >= 64 * 832) return;
    int n = idx / 832, k = idx - n * 832;
    int8_t q = 0;
    if (k < 800) {
        int tap = k >> 5;
        int c   = k & 31;
        float w = w2[(n * 32 + c) * 25 + tap];
        q = (w >= 0.0f) ? (int8_t)1 : (int8_t)-1;
    }
    wb2[idx] = q;
}

__global__ void bnn_conv1(const float* __restrict__ x, const float* __restrict__ w1,
                          const float* __restrict__ sc, const float* __restrict__ sh,
                          int8_t* __restrict__ a1) {
    const int N = 2048 * 13 * 13 * 32;
    int idx = blockIdx.x * blockDim.x + threadIdx.x;
    if (idx >= N) return;
    int c  = idx & 31;
    int t  = idx >> 5;
    int ox = t % 13; t /= 13;
    int oy = t % 13;
    int b  = t / 13;

    float w[9];
#pragma unroll
    for (int i = 0; i < 9; ++i) w[i] = (w1[c * 9 + i] >= 0.0f) ? 1.0f : -1.0f;

    const float* xp = x + b * 784;
    const float scale = sc[c], shift = sh[c];
    int8_t q = -1;
#pragma unroll
    for (int dy = 0; dy < 2; ++dy)
#pragma unroll
        for (int dx = 0; dx < 2; ++dx) {
            int cy = 2 * oy + dy, cx = 2 * ox + dx;
            float acc = 0.0f;
#pragma unroll
            for (int i = 0; i < 3; ++i)
#pragma unroll
                for (int j = 0; j < 3; ++j)
                    acc += xp[(cy + i) * 28 + (cx + j)] * w[i * 3 + j];
            if (acc * scale + shift >= 0.0f) q = 1;
        }
    a1[((b * 13 + oy) * 13 + ox) * 32 + c] = q;
}

__global__ __launch_bounds__(128) void bnn_conv2_wmma(
    const int8_t* __restrict__ a1, const int8_t* __restrict__ wb2,
    const float* __restrict__ sc, const float* __restrict__ sh,
    int8_t* __restrict__ a2) {
    __shared__ int8_t Bs[64 * 832];

#pragma unroll
    for (int it = 0; it < 26; ++it) {
        int ch = threadIdx.x + 128 * it;
        *(int4*)(Bs + ch * 16) = *(const int4*)(wb2 + ch * 16);
    }
    __syncthreads();

    const int lane = threadIdx.x & 31;
    const int wv   = threadIdx.x >> 5;
    const int lm   = lane & 15;
    const int half = lane >> 4;
    const int m_base = (blockIdx.x * 4 + wv) * 16;

    const int m  = m_base + lm;
    const int b  = m / 169;
    int rem      = m - b * 169;
    const int oy = rem / 13;
    const int ox = rem - oy * 13;

    const v8i zero = {};
    v8i acc[4];
#pragma unroll
    for (int j = 0; j < 4; ++j) acc[j] = zero;

    for (int kk = 0; kk < 13; ++kk) {
        const int8_t* q[2];
#pragma unroll
        for (int s = 0; s < 2; ++s) {
            int t = 2 * kk + s;
            const int8_t* p = nullptr;
            if (t < 25) {
                int ky = t / 5, kx = t - (t / 5) * 5;
                int iy = oy + ky - 2, ix = ox + kx - 2;
                if ((unsigned)iy < 13u && (unsigned)ix < 13u)
                    p = a1 + (((b * 13 + iy) * 13 + ix) << 5);
            }
            q[s] = p;
        }
        v8i av;
#pragma unroll
        for (int s = 0; s < 2; ++s) {
            if (q[s]) {
                uint2 d0 = *(const uint2*)(q[s] + half * 16);
                uint2 d1 = *(const uint2*)(q[s] + half * 16 + 8);
                av[4 * s + 0] = (int)d0.x; av[4 * s + 1] = (int)d0.y;
                av[4 * s + 2] = (int)d1.x; av[4 * s + 3] = (int)d1.y;
            } else {
                av[4 * s + 0] = 0; av[4 * s + 1] = 0;
                av[4 * s + 2] = 0; av[4 * s + 3] = 0;
            }
        }
#pragma unroll
        for (int j = 0; j < 4; ++j) {
            const int8_t* bp = Bs + (j * 16 + lm) * 832 + kk * 64 + half * 16;
            int4 t0 = *(const int4*)(bp);
            int4 t1 = *(const int4*)(bp + 32);
            v8i bv;
            bv[0] = t0.x; bv[1] = t0.y; bv[2] = t0.z; bv[3] = t0.w;
            bv[4] = t1.x; bv[5] = t1.y; bv[6] = t1.z; bv[7] = t1.w;
            acc[j] = __builtin_amdgcn_wmma_i32_16x16x64_iu8(
                true, av, true, bv, acc[j], false, false);
        }
    }

#pragma unroll
    for (int j = 0; j < 4; ++j) {
        int n = j * 16 + lm;
        float scale = sc[n], shift = sh[n];
#pragma unroll
        for (int i = 0; i < 8; ++i) {
            int row = m_base + i + half * 8;
            float f = (float)acc[j][i] * scale + shift;
            a2[row * 64 + n] = (f >= 0.0f) ? (int8_t)1 : (int8_t)-1;
        }
    }
}

__global__ void bnn_pool_flatten(const int8_t* __restrict__ a2, int8_t* __restrict__ a3) {
    const int N = 2048 * 2304;
    int idx = blockIdx.x * blockDim.x + threadIdx.x;
    if (idx >= N) return;
    int k = idx % 2304;
    int b = idx / 2304;
    int c = k / 36;
    int r = k - c * 36;
    int y = r / 6, xx = r - y * 6;
    int8_t q = -1;
#pragma unroll
    for (int dy = 0; dy < 2; ++dy)
#pragma unroll
        for (int dx = 0; dx < 2; ++dx) {
            if (a2[((b * 13 + (2 * y + dy)) * 13 + (2 * xx + dx)) * 64 + c] > 0) q = 1;
        }
    a3[idx] = q;
}

__global__ __launch_bounds__(128) void bnn_fc1_wmma(
    const int8_t* __restrict__ a3, const int8_t* __restrict__ wf1p,
    const float* __restrict__ sc, const float* __restrict__ sh,
    int8_t* __restrict__ a4) {
    __shared__ int8_t As[16 * 2320];

    const int m_base = blockIdx.x * 16;

#pragma unroll
    for (int it = 0; it < 18; ++it) {
        int ch  = threadIdx.x + 128 * it;
        int row = ch / 144;
        int col = ch - row * 144;
        *(int4*)(As + row * 2320 + col * 16) =
            *(const int4*)(a3 + (m_base + row) * 2304 + col * 16);
    }
    __syncthreads();

    const int lane = threadIdx.x & 31;
    const int wv   = threadIdx.x >> 5;
    const int lm   = lane & 15;
    const int half = lane >> 4;
    const int n_base = blockIdx.y * 256 + wv * 64;

    const int8_t* arow = As + lm * 2320 + half * 16;

    const v8i zero = {};
    v8i acc[4];
#pragma unroll
    for (int j = 0; j < 4; ++j) acc[j] = zero;

    for (int kk = 0; kk < 36; ++kk) {
        v8i av;
#pragma unroll
        for (int p = 0; p < 4; ++p) {
            uint2 d = *(const uint2*)(arow + kk * 64 + (p >> 1) * 32 + (p & 1) * 8);
            av[2 * p]     = (int)d.x;
            av[2 * p + 1] = (int)d.y;
        }
#pragma unroll
        for (int j = 0; j < 4; ++j) {
            const int8_t* bp = wf1p + (n_base + j * 16 + lm) * 2304 + kk * 64 + half * 16;
            int4 t0 = *(const int4*)(bp);
            int4 t1 = *(const int4*)(bp + 32);
            v8i bv;
            bv[0] = t0.x; bv[1] = t0.y; bv[2] = t0.z; bv[3] = t0.w;
            bv[4] = t1.x; bv[5] = t1.y; bv[6] = t1.z; bv[7] = t1.w;
            acc[j] = __builtin_amdgcn_wmma_i32_16x16x64_iu8(
                true, av, true, bv, acc[j], false, false);
        }
    }

#pragma unroll
    for (int j = 0; j < 4; ++j) {
        int n = n_base + j * 16 + lm;
        float scale = sc[n], shift = sh[n];
#pragma unroll
        for (int i = 0; i < 8; ++i) {
            int row = m_base + i + half * 8;
            float f = (float)acc[j][i] * scale + shift;
            a4[row * 1024 + n] = (f >= 0.0f) ? (int8_t)1 : (int8_t)-1;
        }
    }
}

__device__ __forceinline__ int bnn_dot4(unsigned a, unsigned b) {
    int s = 0;
#pragma unroll
    for (int i = 0; i < 4; ++i) {
        int av = (int)(signed char)(a >> (8 * i));
        int bv = (int)(signed char)(b >> (8 * i));
        s += av * bv;
    }
    return s;
}

__global__ void bnn_fc2(const int8_t* __restrict__ a4, const int8_t* __restrict__ wf2p,
                        const float* __restrict__ sc, const float* __restrict__ sh,
                        float* __restrict__ logits) {
    int idx = blockIdx.x * blockDim.x + threadIdx.x;
    if (idx >= 2048 * 10) return;
    int n = idx % 10, b = idx / 10;
    const int8_t* ap = a4 + b * 1024;
    const int8_t* wp = wf2p + n * 1024;
    int acc = 0;
    for (int k = 0; k < 1024; k += 16) {
        int4 xa = *(const int4*)(ap + k);
        int4 xw = *(const int4*)(wp + k);
        acc += bnn_dot4(xa.x, xw.x) + bnn_dot4(xa.y, xw.y) +
               bnn_dot4(xa.z, xw.z) + bnn_dot4(xa.w, xw.w);
    }
    logits[idx] = (float)acc * sc[n] + sh[n];
}

__global__ void bnn_logsoftmax(const float* __restrict__ logits, float* __restrict__ out) {
    int b = blockIdx.x * blockDim.x + threadIdx.x;
    if (b >= 2048) return;
    float v[10];
    float mx = -3.4e38f;
#pragma unroll
    for (int n = 0; n < 10; ++n) { v[n] = logits[b * 10 + n]; mx = fmaxf(mx, v[n]); }
    float s = 0.0f;
#pragma unroll
    for (int n = 0; n < 10; ++n) s += expf(v[n] - mx);
    float l = mx + logf(s);
#pragma unroll
    for (int n = 0; n < 10; ++n) out[b * 10 + n] = v[n] - l;
}

extern "C" void kernel_launch(void* const* d_in, const int* in_sizes, int n_in,
                              void* d_out, int out_size, void* d_ws, size_t ws_size,
                              hipStream_t stream) {
    (void)in_sizes; (void)n_in; (void)out_size; (void)ws_size;
    const float* x    = (const float*)d_in[0];
    const float* w1   = (const float*)d_in[1];
    const float* b1   = (const float*)d_in[2];
    const float* g1   = (const float*)d_in[3];
    const float* be1  = (const float*)d_in[4];
    const float* m1   = (const float*)d_in[5];
    const float* v1   = (const float*)d_in[6];
    const float* w2   = (const float*)d_in[7];
    const float* b2   = (const float*)d_in[8];
    const float* g2   = (const float*)d_in[9];
    const float* be2  = (const float*)d_in[10];
    const float* m2   = (const float*)d_in[11];
    const float* v2   = (const float*)d_in[12];
    const float* wf1  = (const float*)d_in[13];
    const float* bf1  = (const float*)d_in[14];
    const float* gf1  = (const float*)d_in[15];
    const float* bef1 = (const float*)d_in[16];
    const float* mf1  = (const float*)d_in[17];
    const float* vf1  = (const float*)d_in[18];
    const float* wf2  = (const float*)d_in[19];
    const float* bf2  = (const float*)d_in[20];
    const float* gf2  = (const float*)d_in[21];
    const float* bef2 = (const float*)d_in[22];
    const float* mf2  = (const float*)d_in[23];
    const float* vf2  = (const float*)d_in[24];

    char* ws = (char*)d_ws;
    size_t off = 0;
    auto take = [&](size_t bytes) -> char* {
        char* p = ws + off;
        off += (bytes + 255) & ~(size_t)255;
        return p;
    };
    int8_t* a1    = (int8_t*)take(2048UL * 169 * 32);
    int8_t* a2    = (int8_t*)take(2048UL * 169 * 64);
    int8_t* a3    = (int8_t*)take(2048UL * 2304);
    int8_t* a4    = (int8_t*)take(2048UL * 1024);
    int8_t* wb2   = (int8_t*)take(64UL * 832);
    int8_t* wf1p  = (int8_t*)take(1024UL * 2304);
    int8_t* wf2p  = (int8_t*)take(10UL * 1024);
    float*  lgts  = (float*)take(2048UL * 10 * 4);
    float*  sc1   = (float*)take(32 * 4);
    float*  sh1   = (float*)take(32 * 4);
    float*  sc2   = (float*)take(64 * 4);
    float*  sh2   = (float*)take(64 * 4);
    float*  scf1  = (float*)take(1024 * 4);
    float*  shf1  = (float*)take(1024 * 4);
    float*  scf2  = (float*)take(10 * 4);
    float*  shf2  = (float*)take(10 * 4);

    bnn_prep_affine<<<1, 64, 0, stream>>>(b1, g1, be1, m1, v1, sc1, sh1, 32);
    bnn_prep_affine<<<1, 64, 0, stream>>>(b2, g2, be2, m2, v2, sc2, sh2, 64);
    bnn_prep_affine<<<4, 256, 0, stream>>>(bf1, gf1, bef1, mf1, vf1, scf1, shf1, 1024);
    bnn_prep_affine<<<1, 32, 0, stream>>>(bf2, gf2, bef2, mf2, vf2, scf2, shf2, 10);
    bnn_pack_w2<<<(64 * 832 + 255) / 256, 256, 0, stream>>>(w2, wb2);
    bnn_pack_sign<<<(1024 * 2304 + 255) / 256, 256, 0, stream>>>(wf1, wf1p, 1024 * 2304);
    bnn_pack_sign<<<(10 * 1024 + 255) / 256, 256, 0, stream>>>(wf2, wf2p, 10 * 1024);

    bnn_conv1<<<(2048 * 13 * 13 * 32 + 255) / 256, 256, 0, stream>>>(x, w1, sc1, sh1, a1);
    bnn_conv2_wmma<<<5408, 128, 0, stream>>>(a1, wb2, sc2, sh2, a2);
    bnn_pool_flatten<<<(2048 * 2304 + 255) / 256, 256, 0, stream>>>(a2, a3);
    bnn_fc1_wmma<<<dim3(128, 4), 128, 0, stream>>>(a3, wf1p, scf1, shf1, a4);
    bnn_fc2<<<(2048 * 10 + 255) / 256, 256, 0, stream>>>(a4, wf2p, scf2, shf2, lgts);
    bnn_logsoftmax<<<(2048 + 255) / 256, 256, 0, stream>>>(lgts, (float*)d_out);
}
